// FillNet_68848325754931
// MI455X (gfx1250) — matrix operand order repaired and checked by reading, not compiled, not run on hardware
//
#include <hip/hip_runtime.h>

typedef __attribute__((ext_vector_type(16))) _Float16 v16h;
typedef __attribute__((ext_vector_type(8)))  float    v8f;
typedef __attribute__((ext_vector_type(8)))  unsigned v8u;

#define NQ 65536
#define PP 2048
#define LOG2E 1.4426950408889634f

__global__ __launch_bounds__(256) void rbf_wmma_kernel(
    const int* __restrict__ X, const int* __restrict__ Pat,
    const float* __restrict__ W2, const float* __restrict__ sigmaSq,
    float* __restrict__ out)
{
    // Pattern node data: {px, py, log2e/(2*sigma^2), pad} -> ds_load_b128
    __shared__ float4   sPD[PP];                  // 32 KB
    // B-matrix coefficients: 5 columns x (PP/2) packed f16 pairs.
    // col 0 = 1.0 (denominator), col 1..3 = W2[c-1, p], col 4 = zeros
    // (lanes with D-column >= 4 index the zero column: no cndmask needed).
    __shared__ unsigned sCoef[5 * (PP / 2)];      // 20 KB
    __shared__ float    sEps[8][16][4];           // 2 KB epilogue bounce

    const int tid  = threadIdx.x;
    const int lane = tid & 31;
    const int wave = tid >> 5;
    const int col  = lane & 15;   // D-matrix column owned by this lane
    const int hi   = lane >> 4;   // lane half (WMMA layout)

    // ---- stage pattern data + coefficient table into LDS ----
    for (int p = tid; p < PP; p += 256) {
        float px = (float)Pat[2 * p];
        float py = (float)Pat[2 * p + 1];
        float cs = LOG2E / (2.0f * sigmaSq[p]);   // rbf = exp2(-dsq*cs)
        sPD[p] = make_float4(px, py, cs, 0.0f);
    }
    for (int i = tid; i < 5 * (PP / 2); i += 256) {
        int c  = i / (PP / 2);
        int pp = (i - c * (PP / 2)) * 2;
        float c0 = 0.0f, c1 = 0.0f;
        if (c == 0)      { c0 = 1.0f; c1 = 1.0f; }
        else if (c < 4)  { c0 = W2[(c - 1) * PP + pp];
                           c1 = W2[(c - 1) * PP + pp + 1]; }
        unsigned short h0 = __builtin_bit_cast(unsigned short, (_Float16)c0);
        unsigned short h1 = __builtin_bit_cast(unsigned short, (_Float16)c1);
        sCoef[i] = ((unsigned)h1 << 16) | (unsigned)h0;
    }
    __syncthreads();

    // ---- each wave owns a 16-row query tile ----
    const int tile    = blockIdx.x * 8 + wave;
    const int rowBase = tile * 16;
    const int n       = rowBase + col;           // both lane halves load same 16 rows
    const float xf = (float)X[2 * n];
    const float yf = (float)X[2 * n + 1];

    const int coefBase = ((col < 4) ? col : 4) * (PP / 2);  // col>=4 -> zero column

    v8f acc = {};
    for (int g = 0; g < PP / 32; ++g) {
        const int pBase = g * 32;
        v16h A = {};
        v8u  Bu;
        #pragma unroll
        for (int v = 0; v < 8; ++v) {
            // A-matrix 16x32 f16 layout: lane holds row M=col,
            // K = 2v + (v>=4 ? 8:0) + (hi ? 8:0), and K+1
            const int ka = 2 * v + ((v >= 4) ? 8 : 0) + (hi ? 8 : 0);
            const int p0 = pBase + ka;
            float4 pd0 = sPD[p0];        // broadcast across the 16-lane half
            float4 pd1 = sPD[p0 + 1];
            float dx0 = xf - pd0.x, dy0 = yf - pd0.y;
            float dx1 = xf - pd1.x, dy1 = yf - pd1.y;
            float d0 = __builtin_fmaf(dx0, dx0, dy0 * dy0);
            float d1 = __builtin_fmaf(dx1, dx1, dy1 * dy1);
            float r0 = __builtin_amdgcn_exp2f(-d0 * pd0.z);   // v_exp_f32
            float r1 = __builtin_amdgcn_exp2f(-d1 * pd1.z);
            A[2 * v]     = (_Float16)r0;
            A[2 * v + 1] = (_Float16)r1;

            // B-matrix 32x16 f16 layout: lane holds column N=col,
            // K = 2v + (v>=4 ? 8:0) + (hi ? 8:0), and K+1 (one packed dword)
            const int kb = 2 * v + ((v >= 4) ? 8 : 0) + (hi ? 8 : 0);
            Bu[v] = sCoef[coefBase + ((pBase + kb) >> 1)];
        }
        v16h B = __builtin_bit_cast(v16h, Bu);
        // D[m,0] += den contribution, D[m,1..3] += weighted sums
        acc = __builtin_amdgcn_wmma_f32_16x16x32_f16(
            false, A, false, B, (short)0, acc, false, false);
    }

    // ---- epilogue: D columns 0..3 -> LDS, then divide + store ----
    if (col < 4) {
        #pragma unroll
        for (int j = 0; j < 8; ++j)
            sEps[wave][hi * 8 + j][col] = acc[j];   // VGPR j <-> row (hi*8 + j)
    }
    __syncthreads();
    if (lane < 16) {
        float inv = __builtin_amdgcn_rcpf(sEps[wave][col][0]);  // v_rcp_f32
        int nOut = rowBase + col;
        out[nOut * 3 + 0] = sEps[wave][col][1] * inv;
        out[nOut * 3 + 1] = sEps[wave][col][2] * inv;
        out[nOut * 3 + 2] = sEps[wave][col][3] * inv;
    }
}

extern "C" void kernel_launch(void* const* d_in, const int* in_sizes, int n_in,
                              void* d_out, int out_size, void* d_ws, size_t ws_size,
                              hipStream_t stream) {
    const int*   X    = (const int*)d_in[0];      // [N,2] int32
    const int*   Pat  = (const int*)d_in[1];      // [P,2] int32
    const float* W2   = (const float*)d_in[2];    // [C,P] f32
    const float* sig  = (const float*)d_in[3];    // [P] f32
    float* out = (float*)d_out;                   // [N,3] f32

    // 4096 tiles of 16 rows, 8 waves (tiles) per 256-thread block
    dim3 grid(NQ / (16 * 8));
    rbf_wmma_kernel<<<grid, dim3(256), 0, stream>>>(X, Pat, W2, sig, out);
}
